// ConstraintAwareGNN_76493367542461
// MI455X (gfx1250) — compile-verified, matrix operand order checked by reading, not run on hardware
//
#include <hip/hip_runtime.h>
#include <math.h>

// ---------------- types ----------------
typedef __bf16 bf16_t;
typedef bf16_t v8bf  __attribute__((ext_vector_type(8)));
typedef bf16_t v16bf __attribute__((ext_vector_type(16)));
typedef float  v8f   __attribute__((ext_vector_type(8)));

#define N_NODES  100000
#define N_EDGES  600000
#define EA_EDGES 700000   // edges + self loops
#define N_PROC   192
#define N_GRAPHS 16
#define HID      128
#define EDIM     32

// ---------------- helpers ----------------
__device__ inline void atomicMaxF(float* addr, float v) {
  // works with buffers initialized to -inf (0xFF800000)
  if (v >= 0.f) atomicMax((int*)addr, __float_as_int(v));
  else          atomicMin((unsigned int*)addr, __float_as_uint(v));
}

__global__ void fill_f32(float* __restrict__ p, float v, size_t n) {
  size_t i = (size_t)blockIdx.x * blockDim.x + threadIdx.x;
  if (i < n) p[i] = v;
}

// ---------------- WMMA strip GEMM ----------------
// C[Mr x (NT*16)] = A[Mr x K](bf16 row-major) * Bpacked.
// One wave computes a full 16-row x NT*16-col strip: the A fragment is loaded
// ONCE per K-step and reused across NT WMMAs (A is the streamed/large operand;
// B panels are <=48KB and L2-resident). B is pre-packed to the CDNA5 16-bit
// B-matrix lane layout: one v16bf per lane per (ntile,ktile).
template <int NT>
__global__ __launch_bounds__(256)
void wmma_gemm_strip(const bf16_t* __restrict__ A, const bf16_t* __restrict__ Bp,
                     const float* __restrict__ bias, const float* __restrict__ mask,
                     float* __restrict__ outF, bf16_t* __restrict__ outH,
                     int Mr, int K, int relu)
{
  const int Nc = NT * 16;
  const int KT = K >> 5;
  const int lane = threadIdx.x & 31;
  const int wave = threadIdx.x >> 5;
  int rowT = blockIdx.x * 8 + wave;
  if (rowT >= (Mr >> 4)) return;            // wave-uniform exit: EXEC all-ones for WMMA
  int laneLow = lane & 15, laneHi = lane >> 4;

  // A: lane holds row (rowT*16+laneLow); laneHi selects K-subgroup {0..7,16..23} vs {8..15,24..31}
  const bf16_t* arow = A + (size_t)((rowT << 4) + laneLow) * K + laneHi * 8;

  v8f acc[NT];
  const v8f vzero = {0.f,0.f,0.f,0.f,0.f,0.f,0.f,0.f};
  #pragma unroll
  for (int nt = 0; nt < NT; ++nt) acc[nt] = vzero;

  for (int kt = 0; kt < KT; ++kt) {
    if (kt + 1 < KT) __builtin_prefetch(arow + (kt + 1) * 32, 0, 3);
    v8bf alo = *(const v8bf*)(arow + kt * 32);
    v8bf ahi = *(const v8bf*)(arow + kt * 32 + 16);
    v16bf a = __builtin_shufflevector(alo, ahi, 0,1,2,3,4,5,6,7,8,9,10,11,12,13,14,15);
    const bf16_t* bk = Bp + ((size_t)kt * 32 + lane) * 16;
    #pragma unroll
    for (int nt = 0; nt < NT; ++nt) {
      v16bf b = *(const v16bf*)(bk + (size_t)nt * KT * 512);
      acc[nt] = __builtin_amdgcn_wmma_f32_16x16x32_bf16(false, a, false, b,
                                                        (short)0, acc[nt], false, false);
    }
  }

  #pragma unroll
  for (int nt = 0; nt < NT; ++nt) {
    int col = nt * 16 + laneLow;
    float bv = bias ? bias[col] : 0.f;
    #pragma unroll
    for (int r = 0; r < 8; ++r) {
      int orow = (rowT << 4) + r + laneHi * 8;
      size_t o = (size_t)orow * Nc + col;
      float v = acc[nt][r] + bv;
      if (mask) v += -1e9f * (1.0f - mask[o]);
      if (relu) v = fmaxf(v, 0.f);
      if (outF) outF[o] = v;
      if (outH) outH[o] = (bf16_t)v;
    }
  }
}

// pack f32 weight (K x M row-major; or transpose: src is M x K) into WMMA-B bf16 layout
__global__ void pack_b_bf16(const float* __restrict__ W, bf16_t* __restrict__ out,
                            int K, int M, int transpose)
{
  int idx = blockIdx.x * 256 + threadIdx.x;
  int total = K * M;
  if (idx >= total) return;
  int i    = idx & 15;
  int lane = (idx >> 4) & 31;
  int t    = idx >> 9;               // t = nt*KT + kt
  int KT = K >> 5;
  int kt = t % KT, nt = t / KT;
  int laneLow = lane & 15, laneHi = lane >> 4;
  int n = nt * 16 + laneLow;
  int k = kt * 32 + laneHi * 8 + (i < 8 ? i : i + 8);
  float v = transpose ? W[(size_t)n * K + k] : W[(size_t)k * M + n];
  out[idx] = (bf16_t)v;
}

// ---------------- node encoder: h = relu(LN(x @ node_w + node_b)) -> bf16 ----------------
__global__ __launch_bounds__(128)
void node_enc_kernel(const float* __restrict__ x, const float* __restrict__ W,
                     const float* __restrict__ b, const float* __restrict__ g,
                     const float* __restrict__ beta, bf16_t* __restrict__ h_bf)
{
  __shared__ float sh[8];
  int n = blockIdx.x, j = threadIdx.x;
  float t = b[j] + x[n*3+0]*W[j] + x[n*3+1]*W[HID+j] + x[n*3+2]*W[2*HID+j];
  float s = t, s2 = t * t;
  #pragma unroll
  for (int off = 16; off; off >>= 1) { s += __shfl_xor(s, off); s2 += __shfl_xor(s2, off); }
  int wv = threadIdx.x >> 5;
  if ((threadIdx.x & 31) == 0) { sh[wv] = s; sh[4+wv] = s2; }
  __syncthreads();
  float S = sh[0]+sh[1]+sh[2]+sh[3], S2 = sh[4]+sh[5]+sh[6]+sh[7];
  float mean = S * (1.f/HID);
  float var  = S2 * (1.f/HID) - mean*mean;
  float v = (t - mean) * rsqrtf(var + 1e-5f) * g[j] + beta[j];
  h_bf[(size_t)n*HID + j] = (bf16_t)fmaxf(v, 0.f);
}

// generic relu(LN(in + gbias)) [+ addend] -> f32/bf16
__global__ __launch_bounds__(128)
void ln_relu_kernel(const float* __restrict__ in, const float* __restrict__ gbias,
                    const float* __restrict__ g, const float* __restrict__ beta,
                    const float* __restrict__ addend,
                    float* __restrict__ outF, bf16_t* __restrict__ outH)
{
  __shared__ float sh[8];
  int n = blockIdx.x, j = threadIdx.x;
  size_t o = (size_t)n*HID + j;
  float t = in[o] + gbias[j];
  float s = t, s2 = t * t;
  #pragma unroll
  for (int off = 16; off; off >>= 1) { s += __shfl_xor(s, off); s2 += __shfl_xor(s2, off); }
  int wv = threadIdx.x >> 5;
  if ((threadIdx.x & 31) == 0) { sh[wv] = s; sh[4+wv] = s2; }
  __syncthreads();
  float S = sh[0]+sh[1]+sh[2]+sh[3], S2 = sh[4]+sh[5]+sh[6]+sh[7];
  float mean = S * (1.f/HID);
  float var  = S2 * (1.f/HID) - mean*mean;
  float v = fmaxf((t - mean) * rsqrtf(var + 1e-5f) * g[j] + beta[j], 0.f);
  if (addend) v += addend[o];
  if (outF) outF[o] = v;
  if (outH) outH[o] = (bf16_t)v;
}

// ---------------- platform encoder + proc_emb (192 rows, fused) ----------------
__global__ __launch_bounds__(128)
void plat_enc_kernel(const float* __restrict__ speeds, const int* __restrict__ tiers,
                     const float* __restrict__ locs, const float* __restrict__ W,
                     const float* __restrict__ b, const float* __restrict__ g,
                     const float* __restrict__ beta, const float* __restrict__ procW,
                     const float* __restrict__ procB, float* __restrict__ proc_emb)
{
  __shared__ float sh[8];
  __shared__ float pe[HID];
  int p = blockIdx.x, j = threadIdx.x;
  int tr = tiers[p];
  float pf[7] = { speeds[p], (float)(tr==0), (float)(tr==1), (float)(tr==2),
                  locs[p*3+0], locs[p*3+1], locs[p*3+2] };
  float t = b[j];
  #pragma unroll
  for (int k = 0; k < 7; ++k) t += pf[k] * W[k*HID + j];
  float s = t, s2 = t * t;
  #pragma unroll
  for (int off = 16; off; off >>= 1) { s += __shfl_xor(s, off); s2 += __shfl_xor(s2, off); }
  int wv = threadIdx.x >> 5;
  if ((threadIdx.x & 31) == 0) { sh[wv] = s; sh[4+wv] = s2; }
  __syncthreads();
  float S = sh[0]+sh[1]+sh[2]+sh[3], S2 = sh[4]+sh[5]+sh[6]+sh[7];
  float mean = S * (1.f/HID);
  float var  = S2 * (1.f/HID) - mean*mean;
  pe[j] = fmaxf((t - mean) * rsqrtf(var + 1e-5f) * g[j] + beta[j], 0.f);
  __syncthreads();
  float o = procB[j];
  for (int k = 0; k < HID; ++k) o += pe[k] * procW[k*HID + j];
  proc_emb[(size_t)p*HID + j] = o;
}

// ---------------- edge encoder + self-loop accumulation ----------------
__global__ void edge_enc_kernel(const float* __restrict__ eattr, const float* __restrict__ ew,
                                const float* __restrict__ eb, const int* __restrict__ dst,
                                float* __restrict__ ea_f, bf16_t* __restrict__ ea_bf,
                                float* __restrict__ cnt)
{
  size_t idx = (size_t)blockIdx.x * 256 + threadIdx.x;
  if (idx >= (size_t)N_EDGES * EDIM) return;
  int e = (int)(idx >> 5), j = (int)(idx & 31);
  float v = fmaxf(eattr[e] * ew[j] + eb[j], 0.f);
  ea_f[idx]  = v;
  ea_bf[idx] = (bf16_t)v;
  int d = dst[e];
  atomicAdd(&ea_f[(size_t)(N_EDGES + d) * EDIM + j], v);
  if (j == 0) atomicAdd(&cnt[d], 1.0f);
}

__global__ void loop_div_kernel(float* __restrict__ ea_f, bf16_t* __restrict__ ea_bf,
                                const float* __restrict__ cnt)
{
  size_t idx = (size_t)blockIdx.x * 256 + threadIdx.x;
  if (idx >= (size_t)N_NODES * EDIM) return;
  int n = (int)(idx >> 5);
  size_t o = (size_t)(N_EDGES + n) * EDIM + (idx & 31);
  float v = ea_f[o] / fmaxf(cnt[n], 1.f);
  ea_f[o]  = v;
  ea_bf[o] = (bf16_t)v;
}

// ---------------- GATv2 attention passes (wave per edge, incl. self loops) ----------------
__global__ __launch_bounds__(256)
void gat_alpha1_kernel(const float* __restrict__ xl, const float* __restrict__ xr,
                       const float* __restrict__ EW, const int* __restrict__ src,
                       const int* __restrict__ dst, const float* __restrict__ att,
                       float* __restrict__ alpha, float* __restrict__ amax, int H)
{
  int eid = (blockIdx.x * 256 + threadIdx.x) >> 5;
  int lane = threadIdx.x & 31;
  if (eid >= EA_EDGES) return;
  int s, d;
  if (eid < N_EDGES) { s = src[eid]; d = dst[eid]; } else { s = d = eid - N_EDGES; }
  int G = 32 / H;            // lanes per head
  int head = lane / G;
  const float* xls = xl + (size_t)s*HID;
  const float* xrd = xr + (size_t)d*HID;
  const float* ewr = EW + (size_t)eid*HID;
  float partial = 0.f;
  #pragma unroll
  for (int u = 0; u < 4; ++u) {
    int j = lane*4 + u;
    float m = xls[j] + xrd[j] + ewr[j];
    m = m > 0.f ? m : 0.2f * m;       // leaky relu
    partial += m * att[j];            // att flattened (H, C)
  }
  for (int off = 1; off < G; off <<= 1) partial += __shfl_xor(partial, off);
  if ((lane % G) == 0) {
    alpha[(size_t)eid*H + head] = partial;
    atomicMaxF(&amax[(size_t)d*H + head], partial);
  }
}

__global__ void gat_alpha2_kernel(float* __restrict__ alpha, const float* __restrict__ amax,
                                  float* __restrict__ denom, const int* __restrict__ dst, int H)
{
  int idx = blockIdx.x * 256 + threadIdx.x;
  if (idx >= EA_EDGES * H) return;
  int eid = idx / H, h = idx % H;
  int d = (eid < N_EDGES) ? dst[eid] : eid - N_EDGES;
  float a = expf(alpha[idx] - amax[(size_t)d*H + h]);
  alpha[idx] = a;
  atomicAdd(&denom[(size_t)d*H + h], a);
}

__global__ __launch_bounds__(256)
void gat_scatter_kernel(const float* __restrict__ xl, const float* __restrict__ alpha,
                        const float* __restrict__ denom, const int* __restrict__ src,
                        const int* __restrict__ dst, float* __restrict__ out, int H)
{
  int eid = (blockIdx.x * 256 + threadIdx.x) >> 5;
  int lane = threadIdx.x & 31;
  if (eid >= EA_EDGES) return;
  int s, d;
  if (eid < N_EDGES) { s = src[eid]; d = dst[eid]; } else { s = d = eid - N_EDGES; }
  int G = 32 / H;
  int head = lane / G;
  float w = alpha[(size_t)eid*H + head] / denom[(size_t)d*H + head];
  const float* xs = xl + (size_t)s*HID;
  float* od = out + (size_t)d*HID;
  #pragma unroll
  for (int u = 0; u < 4; ++u) {
    int j = lane*4 + u;
    atomicAdd(&od[j], xs[j] * w);
  }
}

// ---------------- softmax over 192 procs + exp_speed/tp/durations ----------------
__device__ inline float block_reduce_192(float v, float* sh, int is_max)
{
  #pragma unroll
  for (int off = 16; off; off >>= 1) {
    float o = __shfl_xor(v, off);
    v = is_max ? fmaxf(v, o) : v + o;
  }
  int wv = threadIdx.x >> 5;
  __syncthreads();
  if ((threadIdx.x & 31) == 0) sh[wv] = v;
  __syncthreads();
  float r = sh[0];
  for (int w = 1; w < 6; ++w) r = is_max ? fmaxf(r, sh[w]) : r + sh[w];
  return r;
}

__global__ __launch_bounds__(192)
void softmax_kernel(const float* __restrict__ logits, float* __restrict__ probs,
                    const float* __restrict__ speeds, const int* __restrict__ tiers,
                    const float* __restrict__ x, float* __restrict__ tp,
                    float* __restrict__ dur)
{
  __shared__ float sh[6];
  int n = blockIdx.x, j = threadIdx.x;
  size_t o = (size_t)n * N_PROC + j;
  float v = logits[o];
  float m = block_reduce_192(v, sh, 1);
  float e = expf(v - m);
  float S = block_reduce_192(e, sh, 0);
  float p = e / S;
  probs[o] = p;
  float es = block_reduce_192(p * speeds[j], sh, 0);
  int tr = tiers[j];
  float t0 = block_reduce_192(tr == 0 ? p : 0.f, sh, 0);
  float t1 = block_reduce_192(tr == 1 ? p : 0.f, sh, 0);
  float t2 = block_reduce_192(tr == 2 ? p : 0.f, sh, 0);
  if (j == 0) {
    tp[n*3+0] = t0; tp[n*3+1] = t1; tp[n*3+2] = t2;
    float tpt = x[n*3+0];
    dur[n] = fmaxf(tpt / (es + 1e-6f), 0.f) + 0.001f;
  }
}

// ---------------- base = relu(hidden @ st_w2 + st_b2), wave per node ----------------
__global__ __launch_bounds__(256)
void base_kernel(const float* __restrict__ hidden, const float* __restrict__ w2,
                 const float* __restrict__ b2, float* __restrict__ out_base)
{
  int n = (blockIdx.x * 256 + threadIdx.x) >> 5;
  int lane = threadIdx.x & 31;
  if (n >= N_NODES) return;
  const float* hr = hidden + (size_t)n * 64;
  float acc = hr[lane] * w2[lane] + hr[32 + lane] * w2[32 + lane];
  #pragma unroll
  for (int off = 16; off; off >>= 1) acc += __shfl_xor(acc, off);
  if (lane == 0) out_base[n] = fmaxf(acc + b2[0], 0.f);
}

// ---------------- fold tp@cd_w1[288:294] + bias into P/Q ----------------
__global__ void tp_adjust_kernel(float* __restrict__ P, float* __restrict__ Q,
                                 const float* __restrict__ tp, const float* __restrict__ W1,
                                 const float* __restrict__ b1)
{
  size_t idx = (size_t)blockIdx.x * 256 + threadIdx.x;
  if (idx >= (size_t)N_NODES * HID) return;
  int n = (int)(idx >> 7), j = (int)(idx & 127);
  float a = tp[n*3+0], b = tp[n*3+1], c = tp[n*3+2];
  P[idx] += b1[j] + a*W1[288*HID+j] + b*W1[289*HID+j] + c*W1[290*HID+j];
  Q[idx] +=         a*W1[291*HID+j] + b*W1[292*HID+j] + c*W1[293*HID+j];
}

// ---------------- fused comm-delay edge MLP + min_recv segment max ----------------
__global__ __launch_bounds__(256)
void edge_comm_kernel(const float* __restrict__ P, const float* __restrict__ Q,
                      const float* __restrict__ Ecd, const int* __restrict__ src,
                      const int* __restrict__ dst, const float* __restrict__ w2,
                      const float* __restrict__ b2, const float* __restrict__ tp,
                      const float* __restrict__ T, const float* __restrict__ base,
                      const float* __restrict__ dur, float* __restrict__ maxc)
{
  int eid = (blockIdx.x * 256 + threadIdx.x) >> 5;
  int lane = threadIdx.x & 31;
  if (eid >= N_EDGES) return;
  int s = src[eid], d = dst[eid];
  const float* Ps = P + (size_t)s*HID;
  const float* Qd = Q + (size_t)d*HID;
  const float* Ee = Ecd + (size_t)eid*HID;
  float acc = 0.f;
  #pragma unroll
  for (int u = 0; u < 4; ++u) {
    int j = lane*4 + u;
    float z = fmaxf(Ps[j] + Qd[j] + Ee[j], 0.f);
    acc += z * w2[j];
  }
  #pragma unroll
  for (int off = 16; off; off >>= 1) acc += __shfl_xor(acc, off);
  if (lane == 0) {
    float adj = acc + b2[0];
    float tdb = 0.f;
    #pragma unroll
    for (int t = 0; t < 3; ++t)
      #pragma unroll
      for (int u = 0; u < 3; ++u)
        tdb += tp[s*3+t] * T[t*3+u] * tp[d*3+u];
    float comm = fmaxf(tdb + adj, 0.f);
    atomicMaxF(&maxc[d], base[s] + dur[s] + comm);
  }
}

__global__ void finalize_kernel(const float* __restrict__ maxc, const float* __restrict__ base,
                                const float* __restrict__ dur, const int* __restrict__ batch,
                                float* __restrict__ out_start, float* __restrict__ out_endraw,
                                float* __restrict__ out_end, float* __restrict__ out_mk)
{
  int n = blockIdx.x * 256 + threadIdx.x;
  if (n >= N_NODES) return;
  float b = base[n], du = dur[n];
  float mc = maxc[n];
  if (mc == -INFINITY) mc = b;       // no incoming edges
  float st = fmaxf(b, mc);
  out_start[n]  = st;
  out_endraw[n] = b + du;
  float en = st + du;
  out_end[n] = en;
  atomicMaxF(&out_mk[batch[n]], en);
}

// ================= host orchestration =================
extern "C" void kernel_launch(void* const* d_in, const int* in_sizes, int n_in,
                              void* d_out, int out_size, void* d_ws, size_t ws_size,
                              hipStream_t stream)
{
  (void)in_sizes; (void)n_in; (void)out_size; (void)ws_size;
  const float* x        = (const float*)d_in[0];
  const int*   eidx     = (const int*)d_in[1];
  const float* eattr    = (const float*)d_in[2];
  const int*   batch    = (const int*)d_in[3];
  const float* mask     = (const float*)d_in[4];
  const float* speeds   = (const float*)d_in[5];
  const int*   tiers    = (const int*)d_in[6];
  const float* locs     = (const float*)d_in[7];
  const float* node_w   = (const float*)d_in[8];
  const float* node_b   = (const float*)d_in[9];
  const float* node_ln_g= (const float*)d_in[10];
  const float* node_ln_b= (const float*)d_in[11];
  const float* edge_w   = (const float*)d_in[12];
  const float* edge_b   = (const float*)d_in[13];
  const float* plat_w   = (const float*)d_in[14];
  const float* plat_b   = (const float*)d_in[15];
  const float* plat_ln_g= (const float*)d_in[16];
  const float* plat_ln_b= (const float*)d_in[17];
  const float* g1_wl    = (const float*)d_in[18];
  const float* g1_bl    = (const float*)d_in[19];
  const float* g1_wr    = (const float*)d_in[20];
  const float* g1_br    = (const float*)d_in[21];
  const float* g1_att   = (const float*)d_in[22];
  const float* g1_we    = (const float*)d_in[23];
  const float* g1_bias  = (const float*)d_in[24];
  const float* g2_wl    = (const float*)d_in[25];
  const float* g2_bl    = (const float*)d_in[26];
  const float* g2_wr    = (const float*)d_in[27];
  const float* g2_br    = (const float*)d_in[28];
  const float* g2_att   = (const float*)d_in[29];
  const float* g2_we    = (const float*)d_in[30];
  const float* g2_bias  = (const float*)d_in[31];
  const float* bn1_g    = (const float*)d_in[32];
  const float* bn1_b    = (const float*)d_in[33];
  const float* bn2_g    = (const float*)d_in[34];
  const float* bn2_b    = (const float*)d_in[35];
  const float* task_w   = (const float*)d_in[36];
  const float* task_b   = (const float*)d_in[37];
  const float* proc_w   = (const float*)d_in[38];
  const float* proc_b   = (const float*)d_in[39];
  const float* st_w1    = (const float*)d_in[40];
  const float* st_b1    = (const float*)d_in[41];
  const float* st_w2    = (const float*)d_in[42];
  const float* st_b2    = (const float*)d_in[43];
  const float* cd_w1    = (const float*)d_in[44];
  const float* cd_b1    = (const float*)d_in[45];
  const float* cd_w2    = (const float*)d_in[46];
  const float* cd_b2    = (const float*)d_in[47];
  const float* tdelay   = (const float*)d_in[48];

  const int* src = eidx;
  const int* dst = eidx + N_EDGES;

  float* out = (float*)d_out;
  float* out_logits = out;
  float* out_probs  = out_logits + (size_t)N_NODES * N_PROC;
  float* out_base   = out_probs  + (size_t)N_NODES * N_PROC;
  float* out_start  = out_base   + N_NODES;
  float* out_endraw = out_start  + N_NODES;
  float* out_end    = out_endraw + N_NODES;
  float* out_dur    = out_end    + N_NODES;
  float* out_mk     = out_dur    + N_NODES;

  // workspace arena
  char* wsb = (char*)d_ws;
  size_t off = 0;
  auto alloc = [&](size_t bytes) -> void* {
    void* p = wsb + off;
    off = (off + bytes + 255) & ~(size_t)255;
    return p;
  };
  bf16_t* h_bf    = (bf16_t*)alloc((size_t)N_NODES*HID*2);
  bf16_t* h1_bf   = (bf16_t*)alloc((size_t)N_NODES*HID*2);
  bf16_t* hx_bf   = (bf16_t*)alloc((size_t)N_NODES*HID*2);
  bf16_t* task_bf = (bf16_t*)alloc((size_t)N_NODES*HID*2);
  float*  ea_f    = (float*)alloc((size_t)EA_EDGES*EDIM*4);
  bf16_t* ea_bf   = (bf16_t*)alloc((size_t)EA_EDGES*EDIM*2);
  float*  xl      = (float*)alloc((size_t)N_NODES*HID*4);
  float*  xr      = (float*)alloc((size_t)N_NODES*HID*4);
  float*  EW      = (float*)alloc((size_t)EA_EDGES*HID*4);   // reused later as E_cd
  float*  gatout  = (float*)alloc((size_t)N_NODES*HID*4);
  float*  h1_f    = (float*)alloc((size_t)N_NODES*HID*4);
  float*  alpha   = (float*)alloc((size_t)EA_EDGES*4*4);
  float*  amax    = (float*)alloc((size_t)N_NODES*4*4);
  float*  denom   = (float*)alloc((size_t)N_NODES*4*4);
  float*  cnt     = (float*)alloc((size_t)N_NODES*4);
  float*  Pbuf    = (float*)alloc((size_t)N_NODES*HID*4);
  float*  Qbuf    = (float*)alloc((size_t)N_NODES*HID*4);
  float*  hidden  = (float*)alloc((size_t)N_NODES*64*4);
  float*  tp      = (float*)alloc((size_t)N_NODES*3*4);
  float*  maxc    = (float*)alloc((size_t)N_NODES*4);
  float*  proc_emb= (float*)alloc((size_t)N_PROC*HID*4);
  bf16_t* p_wl1 = (bf16_t*)alloc(HID*HID*2);
  bf16_t* p_wr1 = (bf16_t*)alloc(HID*HID*2);
  bf16_t* p_we1 = (bf16_t*)alloc(EDIM*HID*2);
  bf16_t* p_wl2 = (bf16_t*)alloc(HID*HID*2);
  bf16_t* p_wr2 = (bf16_t*)alloc(HID*HID*2);
  bf16_t* p_we2 = (bf16_t*)alloc(EDIM*HID*2);
  bf16_t* p_task= (bf16_t*)alloc(HID*HID*2);
  bf16_t* p_proc= (bf16_t*)alloc(HID*N_PROC*2);
  bf16_t* p_st1 = (bf16_t*)alloc(HID*64*2);
  bf16_t* p_cdA = (bf16_t*)alloc(HID*HID*2);
  bf16_t* p_cdB = (bf16_t*)alloc(HID*HID*2);
  bf16_t* p_cdE = (bf16_t*)alloc(EDIM*HID*2);

  auto fill = [&](float* p, float v, size_t n) {
    fill_f32<<<(unsigned)((n + 255) / 256), 256, 0, stream>>>(p, v, n);
  };
  auto pack = [&](const float* W, bf16_t* dstp, int K, int M, int trans) {
    pack_b_bf16<<<(K*M + 255) / 256, 256, 0, stream>>>(W, dstp, K, M, trans);
  };
  // strip GEMM dispatch: one wave per 16-row strip, full output width
  auto gemm = [&](const bf16_t* A, const bf16_t* Bp, const float* bias, const float* mk,
                  float* oF, bf16_t* oH, int Mr, int Nc, int K, int relu) {
    unsigned blocks = (unsigned)((Mr / 16 + 7) / 8);
    if (Nc == 64)
      wmma_gemm_strip<4><<<blocks, 256, 0, stream>>>(A, Bp, bias, mk, oF, oH, Mr, K, relu);
    else if (Nc == 128)
      wmma_gemm_strip<8><<<blocks, 256, 0, stream>>>(A, Bp, bias, mk, oF, oH, Mr, K, relu);
    else // 192
      wmma_gemm_strip<12><<<blocks, 256, 0, stream>>>(A, Bp, bias, mk, oF, oH, Mr, K, relu);
  };

  // ---- platform encoder + proc_emb, then weight packing ----
  plat_enc_kernel<<<N_PROC, 128, 0, stream>>>(speeds, tiers, locs, plat_w, plat_b,
                                              plat_ln_g, plat_ln_b, proc_w, proc_b, proc_emb);
  pack(g1_wl, p_wl1, HID, HID, 0);
  pack(g1_wr, p_wr1, HID, HID, 0);
  pack(g1_we, p_we1, EDIM, HID, 0);
  pack(g2_wl, p_wl2, HID, HID, 0);
  pack(g2_wr, p_wr2, HID, HID, 0);
  pack(g2_we, p_we2, EDIM, HID, 0);
  pack(task_w, p_task, HID, HID, 0);
  pack(proc_emb, p_proc, HID, N_PROC, 1);          // B = proc_emb^T
  pack(st_w1, p_st1, HID, 64, 0);
  pack(cd_w1,              p_cdA, HID, HID, 0);    // rows 0..127  (hx[src])
  pack(cd_w1 + 128*HID,    p_cdB, HID, HID, 0);    // rows 128..255 (hx[dst])
  pack(cd_w1 + 256*HID,    p_cdE, EDIM, HID, 0);   // rows 256..287 (e)

  // ---- encoders ----
  fill(ea_f + (size_t)N_EDGES*EDIM, 0.f, (size_t)N_NODES*EDIM);
  fill(cnt, 0.f, N_NODES);
  node_enc_kernel<<<N_NODES, 128, 0, stream>>>(x, node_w, node_b, node_ln_g, node_ln_b, h_bf);
  edge_enc_kernel<<<(unsigned)(((size_t)N_EDGES*EDIM + 255)/256), 256, 0, stream>>>(
      eattr, edge_w, edge_b, dst, ea_f, ea_bf, cnt);
  loop_div_kernel<<<(unsigned)(((size_t)N_NODES*EDIM + 255)/256), 256, 0, stream>>>(
      ea_f, ea_bf, cnt);

  // ---- GAT layer 1 (H=4) ----
  gemm(h_bf, p_wl1, g1_bl, nullptr, xl, nullptr, N_NODES, HID, HID, 0);
  gemm(h_bf, p_wr1, g1_br, nullptr, xr, nullptr, N_NODES, HID, HID, 0);
  gemm(ea_bf, p_we1, nullptr, nullptr, EW, nullptr, EA_EDGES, HID, EDIM, 0);
  fill(amax, -INFINITY, (size_t)N_NODES*4);
  fill(denom, 0.f, (size_t)N_NODES*4);
  fill(gatout, 0.f, (size_t)N_NODES*HID);
  gat_alpha1_kernel<<<(unsigned)((EA_EDGES*32 + 255)/256), 256, 0, stream>>>(
      xl, xr, EW, src, dst, g1_att, alpha, amax, 4);
  gat_alpha2_kernel<<<(unsigned)((EA_EDGES*4 + 255)/256), 256, 0, stream>>>(
      alpha, amax, denom, dst, 4);
  gat_scatter_kernel<<<(unsigned)((EA_EDGES*32 + 255)/256), 256, 0, stream>>>(
      xl, alpha, denom, src, dst, gatout, 4);
  ln_relu_kernel<<<N_NODES, 128, 0, stream>>>(gatout, g1_bias, bn1_g, bn1_b,
                                              nullptr, h1_f, h1_bf);

  // ---- GAT layer 2 (H=1) ----
  gemm(h1_bf, p_wl2, g2_bl, nullptr, xl, nullptr, N_NODES, HID, HID, 0);
  gemm(h1_bf, p_wr2, g2_br, nullptr, xr, nullptr, N_NODES, HID, HID, 0);
  gemm(ea_bf, p_we2, nullptr, nullptr, EW, nullptr, EA_EDGES, HID, EDIM, 0);
  fill(amax, -INFINITY, (size_t)N_NODES*4);
  fill(denom, 0.f, (size_t)N_NODES*4);
  fill(gatout, 0.f, (size_t)N_NODES*HID);
  gat_alpha1_kernel<<<(unsigned)((EA_EDGES*32 + 255)/256), 256, 0, stream>>>(
      xl, xr, EW, src, dst, g2_att, alpha, amax, 1);
  gat_alpha2_kernel<<<(unsigned)((EA_EDGES*1 + 255)/256), 256, 0, stream>>>(
      alpha, amax, denom, dst, 1);
  gat_scatter_kernel<<<(unsigned)((EA_EDGES*32 + 255)/256), 256, 0, stream>>>(
      xl, alpha, denom, src, dst, gatout, 1);
  // hx = h1 + relu(ln(gatout + bias)) -> bf16 only
  ln_relu_kernel<<<N_NODES, 128, 0, stream>>>(gatout, g2_bias, bn2_g, bn2_b,
                                              h1_f, nullptr, hx_bf);

  // ---- task head: logits + softmax ----
  gemm(hx_bf, p_task, task_b, nullptr, nullptr, task_bf, N_NODES, HID, HID, 1);
  gemm(task_bf, p_proc, nullptr, mask, out_logits, nullptr, N_NODES, N_PROC, HID, 0);
  softmax_kernel<<<N_NODES, 192, 0, stream>>>(out_logits, out_probs, speeds, tiers,
                                              x, tp, out_dur);

  // ---- base times ----
  gemm(hx_bf, p_st1, st_b1, nullptr, hidden, nullptr, N_NODES, 64, HID, 1);
  base_kernel<<<(unsigned)((N_NODES*32 + 255)/256), 256, 0, stream>>>(hidden, st_w2, st_b2, out_base);

  // ---- comm-delay precomputes (fused decomposition of comb @ cd_w1) ----
  gemm(hx_bf, p_cdA, nullptr, nullptr, Pbuf, nullptr, N_NODES, HID, HID, 0);
  gemm(hx_bf, p_cdB, nullptr, nullptr, Qbuf, nullptr, N_NODES, HID, HID, 0);
  gemm(ea_bf, p_cdE, nullptr, nullptr, EW, nullptr, N_EDGES, HID, EDIM, 0);  // E_cd in EW
  tp_adjust_kernel<<<(unsigned)(((size_t)N_NODES*HID + 255)/256), 256, 0, stream>>>(
      Pbuf, Qbuf, tp, cd_w1, cd_b1);

  // ---- per-edge comm + segment max ----
  fill(maxc, -INFINITY, N_NODES);
  edge_comm_kernel<<<(unsigned)((N_EDGES*32 + 255)/256), 256, 0, stream>>>(
      Pbuf, Qbuf, EW, src, dst, cd_w2, cd_b2, tp, tdelay, out_base, out_dur, maxc);

  // ---- finalize schedule outputs ----
  fill(out_mk, -INFINITY, N_GRAPHS);
  finalize_kernel<<<(unsigned)((N_NODES + 255)/256), 256, 0, stream>>>(
      maxc, out_base, out_dur, batch, out_start, out_endraw, out_end, out_mk);
}
